// AlarmworkRNN_57964878626906
// MI455X (gfx1250) — matrix operand order repaired and checked by reading, not compiled, not run on hardware
//
#include <hip/hip_runtime.h>
#include <hip/hip_bf16.h>
#include <cstdint>
#include <cstddef>

typedef __attribute__((ext_vector_type(16))) __bf16 v16bf;
typedef __attribute__((ext_vector_type(8)))  __bf16 v8bf;
typedef __attribute__((ext_vector_type(8)))  float  v8f;
typedef __attribute__((ext_vector_type(8)))  float  v8fl;

// Problem dims (fixed by the reference).
constexpr int BB = 512;   // batch
constexpr int LL = 512;   // sequence length
constexpr int II = 256;   // input features
constexpr int HH = 1024;  // hidden
constexpr int KC = II + HH;  // fused K dimension = 1280

// ---------- bf16 helpers ----------
__device__ __forceinline__ unsigned short f2bf(float f) {
  unsigned u = __float_as_uint(f);
  unsigned r = (u + 0x7FFFu + ((u >> 16) & 1u)) >> 16;  // round-to-nearest-even
  return (unsigned short)r;
}
__device__ __forceinline__ float bf2f(unsigned short s) {
  return __uint_as_float(((unsigned)s) << 16);
}

// 32-byte fragment container (16 bf16 = one WMMA A or B operand, wave32)
union AB32 {
  uint4 u[2];
  v8bf  h[2];
  unsigned short s[16];
  v16bf v;
};

// ---------- weight concat + bf16 convert: Wc[h][k] = k<I ? W_in[h][k] : W_rec[h][k-I] ----------
__global__ void alarm_wcat_kernel(const float* __restrict__ Win,
                                  const float* __restrict__ Wrec,
                                  unsigned short* __restrict__ Wc) {
  int idx = blockIdx.x * blockDim.x + threadIdx.x;
  if (idx >= HH * KC) return;
  int h = idx / KC, k = idx % KC;
  float v = (k < II) ? Win[h * II + k] : Wrec[h * HH + (k - II)];
  Wc[idx] = f2bf(v);
}

// ---------- zero-init (uint4 granularity) ----------
__global__ void alarm_zero_kernel(uint4* __restrict__ p, int n4) {
  int idx = blockIdx.x * blockDim.x + threadIdx.x;
  if (idx < n4) p[idx] = make_uint4(0u, 0u, 0u, 0u);
}

// 4 WMMAs sharing one A fragment across a 16x64 N-strip
__device__ __forceinline__ void wmma4(const AB32& a,
                                      const unsigned short* __restrict__ Wc,
                                      int bn, int m16, int kh, int kbase,
                                      v8f acc[4]) {
#pragma unroll
  for (int tt = 0; tt < 4; ++tt) {
    const unsigned short* wp =
        Wc + (size_t)(bn + tt * 16 + m16) * KC + kbase + kh * 8;
    AB32 bf;
    bf.u[0] = ((const uint4*)wp)[0];
    bf.u[1] = ((const uint4*)(wp + 16))[0];
    acc[tt] = __builtin_amdgcn_wmma_f32_16x16x32_bf16(
        false, a.v, false, bf.v, (short)0, acc[tt], false, false);
  }
}

// ---------- one recurrent timestep ----------
// grid.x = 256: blocks 0..127 -> layer 1, 128..255 -> layer 2
// block = 128 threads = 4 waves; WG tile 64(M) x 64(N); wave tile 16(M) x 64(N)
__global__ __launch_bounds__(128) void alarm_step_kernel(
    const float* __restrict__ X, int l,
    const unsigned short* __restrict__ Wc1,
    const unsigned short* __restrict__ Wc2,
    const float* __restrict__ b1, const float* __restrict__ b2,
    const unsigned short* __restrict__ z1_cur,
    const unsigned short* __restrict__ z2_cur,
    unsigned short* __restrict__ z1_nxt,
    unsigned short* __restrict__ z2_nxt) {
  const int lane = threadIdx.x & 31;
  const int wv   = threadIdx.x >> 5;
  const int m16  = lane & 15;   // row within 16-tile (A) / column (B,D)
  const int kh   = lane >> 4;   // K-half selector

  const int layer = blockIdx.x >> 7;
  const int t     = blockIdx.x & 127;
  const int bm    = (t >> 4) * 64;  // batch-tile base
  const int bn    = (t & 15) * 64;  // hidden-tile base
  const bool even = ((l & 1) == 0);

  if (layer == 1 && !even) {
    // layer-2 state is frozen on odd steps: propagate tile into next buffer
    for (int i = threadIdx.x; i < 64 * 64 / 8; i += 128) {
      int r = i >> 3, c = i & 7;  // 8 uint4 per 64-col row
      ((uint4*)(z2_nxt + (size_t)(bm + r) * HH + bn))[c] =
          ((const uint4*)(z2_cur + (size_t)(bm + r) * HH + bn))[c];
    }
    return;
  }

  const unsigned short* Wc  = layer ? Wc2 : Wc1;
  const float*          bia = layer ? b2 : b1;
  unsigned short*       dst = layer ? z2_nxt : z1_nxt;

  const int    row  = bm + wv * 16 + m16;              // batch row this lane loads
  const float* xrow = X + ((size_t)row * LL + l) * II; // X[row, l, :]

  v8f acc[4];
#pragma unroll
  for (int tt = 0; tt < 4; ++tt)
#pragma unroll
    for (int e = 0; e < 8; ++e) acc[tt][e] = 0.0f;

  // ---- K over input features (fp32 X, packed-converted on the fly) : kb = 0..7 ----
  for (int kb = 0; kb < II / 32; ++kb) {
    const int kbase = kb * 32;
    __builtin_prefetch(Wc + (size_t)(bn + m16) * KC + kbase + 32, 0, 0);
    const int i0 = kbase + kh * 8;
    v8fl f0 = *(const v8fl*)(xrow + i0);       // 32B aligned
    v8fl f1 = *(const v8fl*)(xrow + i0 + 16);  // 32B aligned
    AB32 a;
    a.h[0] = __builtin_convertvector(f0, v8bf);  // v_cvt_pk_bf16_f32
    a.h[1] = __builtin_convertvector(f1, v8bf);
    wmma4(a, Wc, bn, m16, kh, kbase, acc);
  }

  // ---- K over recurrent state : kb = 8..39 ----
  if (layer == 0) {
    // layer 1 recurrent input is z12 = z1_prev + z2_prev, fused with packed bf16 adds
    for (int kb = II / 32; kb < KC / 32; ++kb) {
      const int kbase = kb * 32;
      const int j0    = kbase - II + kh * 8;
      const v8bf p0 = *(const v8bf*)(z1_cur + (size_t)row * HH + j0);
      const v8bf q0 = *(const v8bf*)(z2_cur + (size_t)row * HH + j0);
      const v8bf p1 = *(const v8bf*)(z1_cur + (size_t)row * HH + j0 + 16);
      const v8bf q1 = *(const v8bf*)(z2_cur + (size_t)row * HH + j0 + 16);
      AB32 a;
      a.h[0] = p0 + q0;   // v_pk_add_bf16
      a.h[1] = p1 + q1;
      wmma4(a, Wc, bn, m16, kh, kbase, acc);
    }
  } else {
    // layer 2 recurrent input is its own state z2_prev
    for (int kb = II / 32; kb < KC / 32; ++kb) {
      const int kbase = kb * 32;
      const int j0    = kbase - II + kh * 8;
      AB32 a;
      a.u[0] = *(const uint4*)(z2_cur + (size_t)row * HH + j0);
      a.u[1] = *(const uint4*)(z2_cur + (size_t)row * HH + j0 + 16);
      wmma4(a, Wc, bn, m16, kh, kbase, acc);
    }
  }

  // ---- bias + tanh + store bf16 ----
#pragma unroll
  for (int tt = 0; tt < 4; ++tt) {
    const int   n  = bn + tt * 16 + m16;
    const float bv = bia[n];
#pragma unroll
    for (int r = 0; r < 8; ++r) {
      const int m  = r + 8 * kh;
      const int rb = bm + wv * 16 + m;
      dst[(size_t)rb * HH + n] = f2bf(tanhf(acc[tt][r] + bv));
    }
  }
}

// ---------- output head: out[b] = tanh(z1 . W_out + b_out) ----------
__global__ void alarm_out_kernel(const unsigned short* __restrict__ z1,
                                 const float* __restrict__ Wout,
                                 const float* __restrict__ bout,
                                 float* __restrict__ out) {
  int b = blockIdx.x * blockDim.x + threadIdx.x;
  if (b >= BB) return;
  float acc = bout[0];
  for (int h = 0; h < HH; ++h) acc += bf2f(z1[(size_t)b * HH + h]) * Wout[h];
  out[b] = tanhf(acc);
}

extern "C" void kernel_launch(void* const* d_in, const int* in_sizes, int n_in,
                              void* d_out, int out_size, void* d_ws, size_t ws_size,
                              hipStream_t stream) {
  (void)in_sizes; (void)n_in; (void)out_size; (void)ws_size;
  const float* X     = (const float*)d_in[0];
  const float* Win1  = (const float*)d_in[1];
  const float* bin1  = (const float*)d_in[2];
  const float* Wrec1 = (const float*)d_in[3];
  const float* Win2  = (const float*)d_in[4];
  const float* bin2  = (const float*)d_in[5];
  const float* Wrec2 = (const float*)d_in[6];
  const float* Wout  = (const float*)d_in[7];
  const float* bout  = (const float*)d_in[8];
  float* out = (float*)d_out;

  // Workspace layout (~9.5 MB total)
  char* ws = (char*)d_ws;
  unsigned short* Wc1 = (unsigned short*)ws;  ws += (size_t)HH * KC * 2;
  unsigned short* Wc2 = (unsigned short*)ws;  ws += (size_t)HH * KC * 2;
  unsigned short* z1b[2];
  unsigned short* z2b[2];
  z1b[0] = (unsigned short*)ws;  ws += (size_t)BB * HH * 2;
  z1b[1] = (unsigned short*)ws;  ws += (size_t)BB * HH * 2;
  z2b[0] = (unsigned short*)ws;  ws += (size_t)BB * HH * 2;
  z2b[1] = (unsigned short*)ws;  ws += (size_t)BB * HH * 2;

  // 1) Concatenate + convert weights to bf16
  {
    int n = HH * KC;
    int g = (n + 255) / 256;
    alarm_wcat_kernel<<<g, 256, 0, stream>>>(Win1, Wrec1, Wc1);
    alarm_wcat_kernel<<<g, 256, 0, stream>>>(Win2, Wrec2, Wc2);
  }
  // 2) Zero initial states (all four ping-pong buffers, contiguous)
  {
    int n4 = (4 * BB * HH * 2) / 16;  // uint4 count
    int g  = (n4 + 255) / 256;
    alarm_zero_kernel<<<g, 256, 0, stream>>>((uint4*)z1b[0], n4);
  }
  // 3) Sequential recurrence: one fused-GEMM kernel per timestep
  for (int l = 0; l < LL; ++l) {
    const unsigned short* z1c = z1b[l & 1];
    const unsigned short* z2c = z2b[l & 1];
    unsigned short* z1n = z1b[(l + 1) & 1];
    unsigned short* z2n = z2b[(l + 1) & 1];
    alarm_step_kernel<<<256, 128, 0, stream>>>(X, l, Wc1, Wc2, bin1, bin2,
                                               z1c, z2c, z1n, z2n);
  }
  // 4) Output head (final z1 lives in buffer parity LL & 1 == 0)
  alarm_out_kernel<<<(BB + 255) / 256, 256, 0, stream>>>(z1b[LL & 1], Wout, bout, out);
}
